// Attention_68401649156342
// MI455X (gfx1250) — compile-verified, matrix operand order checked by reading, not run on hardware
//
#include <hip/hip_runtime.h>
#include <hip/hip_bf16.h>

// ---------------------------------------------------------------------------
// Types for gfx1250 WMMA (wave32)
// ---------------------------------------------------------------------------
typedef __attribute__((ext_vector_type(16))) __bf16 bf16x16;
typedef __attribute__((ext_vector_type(8)))  __bf16 bf16x8;
typedef __attribute__((ext_vector_type(8)))  float  v8f;

#define B_  8
#define S_  2048
#define F_  128
#define D_  256
#define ROWS_TOTAL (B_ * S_)   // 16384

// ---------------------------------------------------------------------------
// Helpers
// ---------------------------------------------------------------------------
__device__ __forceinline__ bf16x16 load_frag(const __bf16* p0, const __bf16* p1) {
    // Two contiguous 16-byte chunks: elements 0..7 <- p0, 8..15 <- p1
    bf16x8 a = *reinterpret_cast<const bf16x8*>(p0);
    bf16x8 b = *reinterpret_cast<const bf16x8*>(p1);
    bf16x16 r;
#pragma unroll
    for (int i = 0; i < 8; ++i) { r[i] = a[i]; r[i + 8] = b[i]; }
    return r;
}

__device__ __forceinline__ v8f wmma_bf16(bf16x16 a, bf16x16 b, v8f c) {
    // D = A(16x32 bf16) x B(32x16 bf16) + C(16x16 f32)
    return __builtin_amdgcn_wmma_f32_16x16x32_bf16(
        /*neg_a=*/false, a, /*neg_b=*/false, b,
        /*c_mod=*/(short)0, c, /*reuse_a=*/false, /*reuse_b=*/false);
}

// CDNA5 async copy: 16 bytes global -> LDS, tracked by ASYNCcnt (GVS mode:
// uniform SGPR base + per-lane 32-bit voffset).
__device__ __forceinline__ void async_ld16(void* lds_ptr, const void* base,
                                           unsigned voff) {
    unsigned lds = (unsigned)(unsigned long long)(uintptr_t)lds_ptr;
    asm volatile("global_load_async_to_lds_b128 %0, %1, %2"
                 :
                 : "v"(lds), "v"(voff), "s"((unsigned long long)(uintptr_t)base)
                 : "memory");
}
__device__ __forceinline__ void wait_async0() {
    asm volatile("s_wait_asynccnt 0" ::: "memory");
}

__device__ __forceinline__ float wave_max32(float v) {
#pragma unroll
    for (int o = 16; o > 0; o >>= 1) v = fmaxf(v, __shfl_xor(v, o, 32));
    return v;
}
__device__ __forceinline__ float wave_sum32(float v) {
#pragma unroll
    for (int o = 16; o > 0; o >>= 1) v += __shfl_xor(v, o, 32);
    return v;
}

// ---------------------------------------------------------------------------
// Kernel 0a: f32 -> bf16 elementwise conversion (x)
// ---------------------------------------------------------------------------
__global__ __launch_bounds__(256)
void cvt_f32_bf16_kernel(const float* __restrict__ in, __bf16* __restrict__ out, int n) {
    int i = blockIdx.x * blockDim.x + threadIdx.x;
    if (i < n) out[i] = (__bf16)in[i];
}

// ---------------------------------------------------------------------------
// Kernel 0b: f32 [K,N] -> bf16 transposed [N,K]  (weights; tiny, done once)
// out[n*K + k] = in[k*N + n]
// ---------------------------------------------------------------------------
__global__ __launch_bounds__(256)
void cvt_transpose_kernel(const float* __restrict__ in, __bf16* __restrict__ out,
                          int K, int N) {
    int i = blockIdx.x * blockDim.x + threadIdx.x;
    if (i < K * N) {
        int n = i / K, k = i - n * K;
        out[i] = (__bf16)in[(size_t)k * N + n];
    }
}

// ---------------------------------------------------------------------------
// Kernel 1: fused 2-layer MLP  out = relu(x@W1+b1)@W2+b2.
// Weights are pre-transposed bf16 (W1t[n*128+k], W2t[n*256+k]) so WMMA
// B-fragments are contiguous 16B global loads (L2-resident).
// One block = 16 rows; 8 waves x 2 N-tiles of 16 = N=256.
// ---------------------------------------------------------------------------
__global__ __launch_bounds__(256)
void mlp_kernel(const __bf16* __restrict__ xbf,
                const __bf16* __restrict__ W1t, const float* __restrict__ b1,
                const __bf16* __restrict__ W2t, const float* __restrict__ b2,
                __bf16* __restrict__ out) {
    __shared__ __align__(16) __bf16 sX[16 * F_];   // 4 KB, x tile row-major
    __shared__ __align__(16) __bf16 sH[16 * D_];   // 8 KB, hidden tile

    const int tid  = threadIdx.x;
    const int w    = tid >> 5;      // wave id (8 waves, wave32)
    const int lane = tid & 31;
    const int r0   = blockIdx.x * 16;

    // async-stage x tile: 16x128 bf16 = 4 KB contiguous = 256 x 16B chunks
    async_ld16(&sX[tid * 8], xbf + (size_t)r0 * F_, (unsigned)(tid * 16));
    wait_async0();
    __syncthreads();

    const int off = (lane >> 4) * 8;   // K-half select per ISA A-layout
    const int m   = lane & 15;
    const int mb  = (lane < 16) ? 0 : 8;
    const int cn0 = w * 32 + (lane & 15);

    // ---- GEMM1: h = relu(x @ W1 + b1), K=128 ----
    v8f acc0 = {}, acc1 = {};
#pragma unroll
    for (int k0 = 0; k0 < F_; k0 += 32) {
        bf16x16 a = load_frag(&sX[m * F_ + k0 + off], &sX[m * F_ + k0 + off + 16]);
        const __bf16* w0 = W1t + (size_t)cn0 * F_ + k0 + off;
        const __bf16* w1 = w0 + (size_t)16 * F_;
        acc0 = wmma_bf16(a, load_frag(w0, w0 + 16), acc0);
        acc1 = wmma_bf16(a, load_frag(w1, w1 + 16), acc1);
    }
    // bias + ReLU -> sH (bf16)
#pragma unroll
    for (int t = 0; t < 2; ++t) {
        v8f acc = t ? acc1 : acc0;
        int cn = cn0 + t * 16;
        float bias = b1[cn];
#pragma unroll
        for (int i = 0; i < 8; ++i)
            sH[(mb + i) * D_ + cn] = (__bf16)fmaxf(acc[i] + bias, 0.0f);
    }
    __syncthreads();

    // ---- GEMM2: out = h @ W2 + b2, K=256 ----
    v8f acc2 = {}, acc3 = {};
#pragma unroll
    for (int k0 = 0; k0 < D_; k0 += 32) {
        bf16x16 a = load_frag(&sH[m * D_ + k0 + off], &sH[m * D_ + k0 + off + 16]);
        const __bf16* w0 = W2t + (size_t)cn0 * D_ + k0 + off;
        const __bf16* w1 = w0 + (size_t)16 * D_;
        acc2 = wmma_bf16(a, load_frag(w0, w0 + 16), acc2);
        acc3 = wmma_bf16(a, load_frag(w1, w1 + 16), acc3);
    }
    // bias + store bf16
#pragma unroll
    for (int t = 0; t < 2; ++t) {
        v8f acc = t ? acc3 : acc2;
        int cn = cn0 + t * 16;
        float bias = b2[cn];
#pragma unroll
        for (int i = 0; i < 8; ++i)
            out[(size_t)(r0 + mb + i) * D_ + cn] = (__bf16)(acc[i] + bias);
    }
}

// ---------------------------------------------------------------------------
// Kernel 2: fused Q@K^T + row-max normalization + softmax.
// One block = 32 query rows (two 16-row M-tiles) x all 2048 key columns, so
// every K B-fragment fetched from global feeds 2 WMMAs. A-fragments are
// re-read from LDS each N-tile (b128 LDS loads co-issue under WMMA latency;
// keeping them in registers caused spills). Logits panel 32x2048 f32 (256 KB)
// + Q tile (16 KB) live in dynamic LDS (gfx1250: 320 KB/WGP) -> 1 block/WGP,
// so amdgpu_waves_per_eu(2) hands the allocator the full register budget.
// ---------------------------------------------------------------------------
__global__ __launch_bounds__(256)
__attribute__((amdgpu_waves_per_eu(2)))
void score_softmax_kernel(const __bf16* __restrict__ Qbf,
                          const __bf16* __restrict__ Kbf,
                          float* __restrict__ out) {
    extern __shared__ __align__(16) char smem[];
    float*  L  = reinterpret_cast<float*>(smem);                   // 32 x 2048 f32
    __bf16* sQ = reinterpret_cast<__bf16*>(smem + 32 * S_ * 4);    // 32 x 256 bf16

    const int tid  = threadIdx.x;
    const int w    = tid >> 5;
    const int lane = tid & 31;
    const int b    = blockIdx.x >> 6;          // 64 blocks per batch
    const int r0   = (blockIdx.x & 63) << 5;   // 32 rows per block

    // async-stage Q tile: 32x256 bf16 = 16 KB contiguous = 1024 x 16B chunks
    const __bf16* Qbase = Qbf + ((size_t)b * S_ + r0) * D_;
#pragma unroll
    for (int c = 0; c < 4; ++c)
        async_ld16(&sQ[(tid + c * 256) * 8], Qbase, (unsigned)((tid + c * 256) * 16));
    wait_async0();
    __syncthreads();

    const int off = (lane >> 4) * 8;
    const int m   = lane & 15;
    const int mb  = (lane < 16) ? 0 : 8;

    // each wave owns a contiguous 256-column span: 16 N-tiles of 16
    for (int nt = 0; nt < 16; ++nt) {
        const int t0 = w * 256 + nt * 16;
        const __bf16* Kb = Kbf + ((size_t)b * S_ + t0 + (lane & 15)) * D_;
        v8f acc0 = {}, acc1 = {};
#pragma unroll
        for (int k = 0; k < 8; ++k) {
            const __bf16* p0 = &sQ[m * D_ + k * 32 + off];
            const __bf16* p1 = &sQ[(m + 16) * D_ + k * 32 + off];
            bf16x16 bfr = load_frag(Kb + k * 32 + off, Kb + k * 32 + off + 16);
            acc0 = wmma_bf16(load_frag(p0, p0 + 16), bfr, acc0);
            acc1 = wmma_bf16(load_frag(p1, p1 + 16), bfr, acc1);
        }
        const int col = t0 + (lane & 15);
#pragma unroll
        for (int i = 0; i < 8; ++i) {
            L[(mb + i) * S_ + col]      = acc0[i];
            L[(16 + mb + i) * S_ + col] = acc1[i];
        }
    }
    __syncthreads();

    // row-max normalization + stable softmax; each wave handles 4 rows
    for (int rr = 0; rr < 4; ++rr) {
        const int r = w * 4 + rr;
        float* Lr = &L[r * S_];

        float mx = -3.402823e38f;
        for (int c = lane; c < S_; c += 32) mx = fmaxf(mx, Lr[c]);
        mx = wave_max32(mx);
        const float inv_rmax = 1.0f / mx;          // TEMPERATURE == 1.0

        float zm = -3.402823e38f;
        for (int c = lane; c < S_; c += 32) zm = fmaxf(zm, Lr[c] * inv_rmax);
        zm = wave_max32(zm);

        float s = 0.0f;
        for (int c = lane; c < S_; c += 32) {
            float e = __expf(Lr[c] * inv_rmax - zm);
            Lr[c] = e;                 // cache exp, avoid 2nd transcendental pass
            s += e;
        }
        s = wave_sum32(s);
        const float inv = 1.0f / s;

        float* Or = out + ((size_t)b * S_ + r0 + r) * S_;
        for (int c = lane; c < S_; c += 32)
            Or[c] = Lr[c] * inv;
    }
}

// ---------------------------------------------------------------------------
// Launch
// ---------------------------------------------------------------------------
extern "C" void kernel_launch(void* const* d_in, const int* in_sizes, int n_in,
                              void* d_out, int out_size, void* d_ws, size_t ws_size,
                              hipStream_t stream) {
    const float* x   = (const float*)d_in[0];
    const float* Wq1 = (const float*)d_in[1];
    const float* bq1 = (const float*)d_in[2];
    const float* Wq2 = (const float*)d_in[3];
    const float* bq2 = (const float*)d_in[4];
    const float* Wk1 = (const float*)d_in[5];
    const float* bk1 = (const float*)d_in[6];
    const float* Wk2 = (const float*)d_in[7];
    const float* bk2 = (const float*)d_in[8];
    float* out = (float*)d_out;

    // workspace layout (bytes)
    char* ws = (char*)d_ws;
    __bf16* xbf   = (__bf16*)(ws);                         // 16384*128
    __bf16* wq1t  = (__bf16*)(ws + 4194304);               // 256*128 (transposed)
    __bf16* wq2t  = (__bf16*)(ws + 4259840);               // 256*256 (transposed)
    __bf16* wk1t  = (__bf16*)(ws + 4390912);
    __bf16* wk2t  = (__bf16*)(ws + 4456448);
    __bf16* Qbf   = (__bf16*)(ws + 4587520);               // 16384*256
    __bf16* Kbf   = (__bf16*)(ws + 12976128);              // 16384*256

    // 1) f32 -> bf16 conversions (+ weight transposes, done once, tiny)
    {
        int n = ROWS_TOTAL * F_;   // 2,097,152
        cvt_f32_bf16_kernel<<<(n + 255) / 256, 256, 0, stream>>>(x, xbf, n);
        n = F_ * D_;
        cvt_transpose_kernel<<<(n + 255) / 256, 256, 0, stream>>>(Wq1, wq1t, F_, D_);
        cvt_transpose_kernel<<<(n + 255) / 256, 256, 0, stream>>>(Wk1, wk1t, F_, D_);
        n = D_ * D_;
        cvt_transpose_kernel<<<(n + 255) / 256, 256, 0, stream>>>(Wq2, wq2t, D_, D_);
        cvt_transpose_kernel<<<(n + 255) / 256, 256, 0, stream>>>(Wk2, wk2t, D_, D_);
    }

    // 2) fused MLPs -> Q, K (bf16)
    mlp_kernel<<<ROWS_TOTAL / 16, 256, 0, stream>>>(xbf, wq1t, bq1, wq2t, bq2, Qbf);
    mlp_kernel<<<ROWS_TOTAL / 16, 256, 0, stream>>>(xbf, wk1t, bk1, wk2t, bk2, Kbf);

    // 3) fused scores + softmax
    //    dynamic LDS: 32*2048*4 + 32*256*2 = 278528 B (< 320 KB/WGP)
    const int smem_bytes = 32 * S_ * 4 + 32 * D_ * 2;
    hipFuncSetAttribute(reinterpret_cast<const void*>(&score_softmax_kernel),
                        hipFuncAttributeMaxDynamicSharedMemorySize, smem_bytes);
    score_softmax_kernel<<<B_ * (S_ / 32), 256, smem_bytes, stream>>>(Qbf, Kbf, out);
}
